// BidirectionalAttention_28424093565491
// MI455X (gfx1250) — matrix operand order repaired and checked by reading, not compiled, not run on hardware
//
#include <hip/hip_runtime.h>
#include <hip/hip_bf16.h>
#include <cstddef>

// ---------------------------------------------------------------------------
// CDNA5 (gfx1250) types / helpers
// ---------------------------------------------------------------------------
typedef __attribute__((ext_vector_type(16))) __bf16 v16bf;
typedef __attribute__((ext_vector_type(8)))  float  v8f;

union FragBF {
    unsigned int u[8];
    v16bf v;
};

__device__ __forceinline__ unsigned short f32_to_bf16(float f) {
    unsigned int u = __float_as_uint(f);
    u += 0x7fffu + ((u >> 16) & 1u);   // round-to-nearest-even
    return (unsigned short)(u >> 16);
}

__device__ __forceinline__ v8f wmma_bf16(const FragBF& a, const FragBF& b, v8f c) {
    // v_wmma_f32_16x16x32_bf16 : D = A(16x32) * B(32x16) + C(16x16)
    return __builtin_amdgcn_wmma_f32_16x16x32_bf16(
        /*neg_a=*/false, a.v, /*neg_b=*/false, b.v,
        /*c_mod=*/(short)0, c, /*reuse_a=*/false, /*reuse_b=*/false);
}

// ---- async global->LDS staging (CDNA5 GLOBAL_LOAD_ASYNC_TO_LDS_B128) -------
#if defined(__has_builtin)
#if __has_builtin(__builtin_amdgcn_global_load_async_to_lds_b128)
#define HAS_ASYNC_LDS 1
#endif
#endif
#ifndef HAS_ASYNC_LDS
#define HAS_ASYNC_LDS 0
#endif

#if HAS_ASYNC_LDS
typedef __attribute__((ext_vector_type(4))) int v4i_;
typedef __attribute__((address_space(1))) v4i_* gv4p;   // global int4*
typedef __attribute__((address_space(3))) v4i_* lv4p;   // LDS int4*
#endif

__device__ __forceinline__ void async_wait_all() {
#if __has_builtin(__builtin_amdgcn_s_wait_asynccnt)
    __builtin_amdgcn_s_wait_asynccnt(0);
#else
    asm volatile("s_wait_asynccnt 0x0" ::: "memory");
#endif
}

// ---------------------------------------------------------------------------
// fp32 -> bf16 elementwise conversion (bandwidth-bound prologue)
// ---------------------------------------------------------------------------
__global__ void cvt_f32_bf16(const float* __restrict__ in,
                             unsigned short* __restrict__ out, int n) {
    int i = blockIdx.x * blockDim.x + threadIdx.x;
    if (i < n) out[i] = f32_to_bf16(in[i]);
}

// ---------------------------------------------------------------------------
// bf16 WMMA GEMM:  C[M][N] (f32) = A[M][K] (bf16) * B[K][N] (bf16)
// macro-tile 128x64, 256 threads = 8 waves (4x2), each wave 32x32 (2x2 WMMA).
// A tile staged via async global->LDS when available.
// ---------------------------------------------------------------------------
__global__ __launch_bounds__(256)
void gemm_bf16(const unsigned short* __restrict__ A,
               const unsigned short* __restrict__ Bm,
               float* __restrict__ C,
               int M, int N, int K)
{
    __shared__ unsigned short As[128][32];   // A tile, row-major
    __shared__ unsigned short BsT[64][32];   // B tile, transposed (n-major)

    const int tid  = threadIdx.x;
    const int lane = tid & 31;
    const int wave = tid >> 5;
    const int wm   = wave >> 1;   // 0..3
    const int wn   = wave & 1;    // 0..1
    const int m0   = blockIdx.y * 128;
    const int n0   = blockIdx.x * 64;
    const int lm   = lane & 15;
    const int kh   = lane >> 4;

    v8f zero = {0.f,0.f,0.f,0.f,0.f,0.f,0.f,0.f};
    v8f acc[2][2];
    acc[0][0] = zero; acc[0][1] = zero; acc[1][0] = zero; acc[1][1] = zero;

    const int arow  = tid >> 1;          // 0..127
    const int ahalf = (tid & 1) * 16;    // 0 / 16
    const int brow  = tid >> 3;          // 0..31
    const int bcol  = (tid & 7) * 8;     // 0..56

    for (int k0 = 0; k0 < K; k0 += 32) {
        // stage A tile (each thread: 16 bf16 = 32B)
        {
            const unsigned short* src = A + (size_t)(m0 + arow) * K + k0 + ahalf;
#if HAS_ASYNC_LDS
            __builtin_amdgcn_global_load_async_to_lds_b128(
                (gv4p)src, (lv4p)&As[arow][ahalf], 0, 0);
            __builtin_amdgcn_global_load_async_to_lds_b128(
                (gv4p)src, (lv4p)&As[arow][ahalf], 16, 0);
#else
            const uint4* s4 = (const uint4*)src;
            uint4 d0 = s4[0], d1 = s4[1];
            uint4* dst = (uint4*)&As[arow][ahalf];
            dst[0] = d0; dst[1] = d1;
#endif
        }
        // stage B tile transposed (each thread: 8 bf16, scatter by column)
        {
            union { uint4 q; unsigned short s[8]; } u;
            u.q = *(const uint4*)(Bm + (size_t)(k0 + brow) * N + n0 + bcol);
            #pragma unroll
            for (int j = 0; j < 8; ++j) BsT[bcol + j][brow] = u.s[j];
        }
        if (k0 + 32 < K)
            __builtin_prefetch(A + (size_t)(m0 + arow) * K + k0 + 32 + ahalf, 0, 1);
#if HAS_ASYNC_LDS
        async_wait_all();
#endif
        __syncthreads();

        FragBF a[2], b[2];
        // A fragment: lane holds row M=lm; K contiguous k = 16*kh + s (matches B)
        #pragma unroll
        for (int i = 0; i < 2; ++i) {
            const unsigned short* row = &As[wm * 32 + i * 16 + lm][0];
            #pragma unroll
            for (int j = 0; j < 8; ++j) {
                int kk = (kh << 4) + (j << 1);
                a[i].u[j] = *(const unsigned int*)(row + kk);
            }
        }
        // B fragment: lane holds col N=lm; K contiguous k = 16*kh + s (matches A)
        #pragma unroll
        for (int i = 0; i < 2; ++i) {
            const unsigned short* row = &BsT[wn * 32 + i * 16 + lm][0];
            #pragma unroll
            for (int j = 0; j < 8; ++j) {
                int kk = (kh << 4) + (j << 1);
                b[i].u[j] = *(const unsigned int*)(row + kk);
            }
        }
        #pragma unroll
        for (int i = 0; i < 2; ++i)
            #pragma unroll
            for (int j = 0; j < 2; ++j)
                acc[i][j] = wmma_bf16(a[i], b[j], acc[i][j]);
        __syncthreads();
    }

    // C layout: VGPR r -> M = r + 8*kh ; lane low nibble -> N
    #pragma unroll
    for (int i = 0; i < 2; ++i)
        #pragma unroll
        for (int j = 0; j < 2; ++j)
            #pragma unroll
            for (int r = 0; r < 8; ++r) {
                int m = m0 + wm * 32 + i * 16 + r + 8 * kh;
                int n = n0 + wn * 32 + j * 16 + lm;
                C[(size_t)m * N + n] = acc[i][j][r];
            }
}

// ---------------------------------------------------------------------------
// Fused RMSNorm (+weight) + RoPE, fp32 in -> bf16 out. One block per (b,t).
// cols = 2048 (Q, 16 heads) or 512 (K, 4 heads); head_dim = 128, half = 64.
// ---------------------------------------------------------------------------
__global__ __launch_bounds__(256)
void rmsnorm_rope(const float* __restrict__ X,
                  const float* __restrict__ cosb,
                  const float* __restrict__ sinb,
                  const float* __restrict__ w,
                  unsigned short* __restrict__ Y,
                  int cols, int heads)
{
    const int row = blockIdx.x;           // b*T + t
    const int t = row & 2047;
    const float* x = X + (size_t)row * cols;

    float ss = 0.f;
    for (int i = threadIdx.x; i < cols; i += 256) { float v = x[i]; ss += v * v; }
    #pragma unroll
    for (int d = 1; d < 32; d <<= 1) ss += __shfl_xor(ss, d, 32);
    __shared__ float red[8];
    if ((threadIdx.x & 31) == 0) red[threadIdx.x >> 5] = ss;
    __syncthreads();
    float tot = 0.f;
    #pragma unroll
    for (int i = 0; i < 8; ++i) tot += red[i];
    const float rn = rsqrtf(tot / (float)cols + 1e-5f);

    unsigned short* y = Y + (size_t)row * cols;
    const int pairs = heads * 64;
    for (int p = threadIdx.x; p < pairs; p += 256) {
        int hh = p >> 6, j = p & 63;
        int i1 = hh * 128 + j, i2 = i1 + 64;
        float x1 = x[i1] * rn * w[i1];
        float x2 = x[i2] * rn * w[i2];
        float c = cosb[t * 64 + j], s = sinb[t * 64 + j];
        y[i1] = f32_to_bf16(x1 * c + x2 * s);
        y[i2] = f32_to_bf16(x2 * c - x1 * s);
    }
}

// ---------------------------------------------------------------------------
// Flash attention (bidirectional, GQA 4:1). bf16 Q/K/V [B,T,H,D], bf16 out.
// Block = 128 threads = 4 waves; each wave owns 16 query rows; grid (T/64, B*16).
// K tile staged via async global->LDS when available; V staged transposed.
// ---------------------------------------------------------------------------
__global__ __launch_bounds__(128)
void attn_wmma(const unsigned short* __restrict__ Q,
               const unsigned short* __restrict__ Kb,
               const unsigned short* __restrict__ Vb,
               unsigned short* __restrict__ Y)
{
    constexpr int T = 2048, HD = 128, NH = 16, NKV = 4;
    const int bh  = blockIdx.y;
    const int b   = bh >> 4, h = bh & 15;
    const int kvh = h >> 2;
    const int tid  = threadIdx.x;
    const int lane = tid & 31;
    const int wave = tid >> 5;
    const int lm = lane & 15;
    const int kh = lane >> 4;
    const int q0 = blockIdx.x * 64 + wave * 16;

    __shared__ unsigned short Ks [32][128];   // K tile   [key][dim]
    __shared__ unsigned short VsT[128][32];   // V tile^T [dim][key]
    __shared__ unsigned short Ps [4][16][32]; // per-wave P tile [query][key]

    // preload Q fragments (A layout): 4 chunks of head_dim 32
    FragBF aq[4];
    {
        const unsigned short* qrow = Q + ((size_t)(b * T + q0 + lm) * NH + h) * HD;
        #pragma unroll
        for (int c = 0; c < 4; ++c)
            #pragma unroll
            for (int j = 0; j < 8; ++j) {
                int kk = c * 32 + (kh << 4) + (j << 1);
                aq[c].u[j] = *(const unsigned int*)(qrow + kk);
            }
    }

    v8f zero = {0.f,0.f,0.f,0.f,0.f,0.f,0.f,0.f};
    float mrun[8], lrun[8];
    v8f acc[8];
    #pragma unroll
    for (int r = 0; r < 8; ++r) { mrun[r] = -1e30f; lrun[r] = 0.f; acc[r] = zero; }

    const float scale = 0.08838834764831845f;  // 1/sqrt(128)
    const int ckey = tid >> 2;                 // 0..31
    const int cdim = (tid & 3) * 32;           // 0,32,64,96

    for (int kb0 = 0; kb0 < T; kb0 += 32) {
        // cooperatively stage K tile (async when available) + transposed V tile
        {
            const unsigned short* ksrc =
                Kb + ((size_t)(b * T + kb0 + ckey) * NKV + kvh) * HD + cdim;
#if HAS_ASYNC_LDS
            __builtin_amdgcn_global_load_async_to_lds_b128(
                (gv4p)ksrc, (lv4p)&Ks[ckey][cdim], 0, 0);
            __builtin_amdgcn_global_load_async_to_lds_b128(
                (gv4p)ksrc, (lv4p)&Ks[ckey][cdim], 16, 0);
            __builtin_amdgcn_global_load_async_to_lds_b128(
                (gv4p)ksrc, (lv4p)&Ks[ckey][cdim], 32, 0);
            __builtin_amdgcn_global_load_async_to_lds_b128(
                (gv4p)ksrc, (lv4p)&Ks[ckey][cdim], 48, 0);
#else
            uint4* d = (uint4*)&Ks[ckey][cdim];
            const uint4* s = (const uint4*)ksrc;
            d[0] = s[0]; d[1] = s[1]; d[2] = s[2]; d[3] = s[3];
#endif
            const unsigned short* vsrc =
                Vb + ((size_t)(b * T + kb0 + ckey) * NKV + kvh) * HD + cdim;
            #pragma unroll
            for (int j = 0; j < 32; ++j) VsT[cdim + j][ckey] = vsrc[j];
        }
#if HAS_ASYNC_LDS
        async_wait_all();
#endif
        __syncthreads();

        // S = Q * K^T for 32 keys (two 16-key WMMA tiles, 4 head-dim chunks each)
        v8f s0 = zero, s1 = zero;
        FragBF bk;
        #pragma unroll
        for (int c = 0; c < 4; ++c) {
            const unsigned short* row = &Ks[lm][0];
            #pragma unroll
            for (int j = 0; j < 8; ++j) {
                int kk = c * 32 + (kh << 4) + (j << 1);
                bk.u[j] = *(const unsigned int*)(row + kk);
            }
            s0 = wmma_bf16(aq[c], bk, s0);
        }
        #pragma unroll
        for (int c = 0; c < 4; ++c) {
            const unsigned short* row = &Ks[16 + lm][0];
            #pragma unroll
            for (int j = 0; j < 8; ++j) {
                int kk = c * 32 + (kh << 4) + (j << 1);
                bk.u[j] = *(const unsigned int*)(row + kk);
            }
            s1 = wmma_bf16(aq[c], bk, s1);
        }

        // online softmax: row m = r + 8*kh, key n = lm (tile0) / 16+lm (tile1)
        #pragma unroll
        for (int r = 0; r < 8; ++r) {
            float x0 = s0[r] * scale;
            float x1 = s1[r] * scale;
            float mx = fmaxf(x0, x1);
            #pragma unroll
            for (int d = 1; d < 16; d <<= 1) mx = fmaxf(mx, __shfl_xor(mx, d, 32));
            float mnew  = fmaxf(mrun[r], mx);
            float alpha = __expf(mrun[r] - mnew);
            float p0 = __expf(x0 - mnew);
            float p1 = __expf(x1 - mnew);
            float rs = p0 + p1;
            #pragma unroll
            for (int d = 1; d < 16; d <<= 1) rs += __shfl_xor(rs, d, 32);
            lrun[r] = lrun[r] * alpha + rs;
            mrun[r] = mnew;
            #pragma unroll
            for (int d = 0; d < 8; ++d) acc[d][r] *= alpha;
            Ps[wave][r + 8 * kh][lm]      = f32_to_bf16(p0);
            Ps[wave][r + 8 * kh][16 + lm] = f32_to_bf16(p1);
        }

        // O += P * V : A = P (16x32 keys), B = V^T chunks (32 keys x 16 dims)
        FragBF ap, bv;
        #pragma unroll
        for (int j = 0; j < 8; ++j) {
            int kk = (kh << 4) + (j << 1);
            ap.u[j] = *(const unsigned int*)&Ps[wave][lm][kk];
        }
        #pragma unroll
        for (int d = 0; d < 8; ++d) {
            #pragma unroll
            for (int j = 0; j < 8; ++j) {
                int kk = (kh << 4) + (j << 1);
                bv.u[j] = *(const unsigned int*)&VsT[d * 16 + lm][kk];
            }
            acc[d] = wmma_bf16(ap, bv, acc[d]);
        }
        __syncthreads();
    }

    // epilogue: normalize by l and emit bf16 y [B,T,H,D]
    #pragma unroll
    for (int r = 0; r < 8; ++r) {
        float inv = 1.f / lrun[r];
        unsigned short* dst =
            Y + ((size_t)(b * T + q0 + r + 8 * kh) * NH + h) * HD;
        #pragma unroll
        for (int d = 0; d < 8; ++d)
            dst[d * 16 + lm] = f32_to_bf16(acc[d][r] * inv);
    }
}

// ---------------------------------------------------------------------------
// Launcher
// ---------------------------------------------------------------------------
extern "C" void kernel_launch(void* const* d_in, const int* in_sizes, int n_in,
                              void* d_out, int out_size, void* d_ws, size_t ws_size,
                              hipStream_t stream)
{
    (void)in_sizes; (void)n_in; (void)out_size; (void)ws_size;

    const float* x    = (const float*)d_in[0];
    const float* cosb = (const float*)d_in[1];
    const float* sinb = (const float*)d_in[2];
    const float* wq   = (const float*)d_in[3];
    const float* wk   = (const float*)d_in[4];
    const float* wv   = (const float*)d_in[5];
    const float* wo   = (const float*)d_in[6];
    const float* qw   = (const float*)d_in[7];
    const float* kw   = (const float*)d_in[8];
    float* out = (float*)d_out;

    const int Bc = 4, Tc = 2048, Cc = 2048, KV = 512;
    const int M = Bc * Tc;  // 8192

    char* p = (char*)d_ws;
    auto alloc = [&](size_t bytes) -> char* {
        char* r = p; p += (bytes + 255) & ~(size_t)255; return r;
    };
    unsigned short* x_bf  = (unsigned short*)alloc((size_t)M  * Cc * 2);
    unsigned short* wq_bf = (unsigned short*)alloc((size_t)Cc * Cc * 2);
    unsigned short* wk_bf = (unsigned short*)alloc((size_t)Cc * KV * 2);
    unsigned short* wv_bf = (unsigned short*)alloc((size_t)Cc * KV * 2);
    unsigned short* wo_bf = (unsigned short*)alloc((size_t)Cc * Cc * 2);
    float*          q_f   = (float*)alloc((size_t)M * Cc * 4);
    float*          k_f   = (float*)alloc((size_t)M * KV * 4);
    float*          v_f   = (float*)alloc((size_t)M * KV * 4);
    unsigned short* q_bf  = (unsigned short*)alloc((size_t)M * Cc * 2);
    unsigned short* k_bf  = (unsigned short*)alloc((size_t)M * KV * 2);
    unsigned short* v_bf  = (unsigned short*)alloc((size_t)M * KV * 2);
    unsigned short* y_bf  = (unsigned short*)alloc((size_t)M * Cc * 2);

    auto cvt = [&](const float* src, unsigned short* dst, int n) {
        cvt_f32_bf16<<<(n + 255) / 256, 256, 0, stream>>>(src, dst, n);
    };

    // 1) bf16 conversions
    cvt(x,  x_bf,  M * Cc);
    cvt(wq, wq_bf, Cc * Cc);
    cvt(wk, wk_bf, Cc * KV);
    cvt(wv, wv_bf, Cc * KV);
    cvt(wo, wo_bf, Cc * Cc);

    // 2) QKV projections (bf16 WMMA, fp32 accumulate)
    gemm_bf16<<<dim3(Cc / 64, M / 128), 256, 0, stream>>>(x_bf, wq_bf, q_f, M, Cc, Cc);
    gemm_bf16<<<dim3(KV / 64, M / 128), 256, 0, stream>>>(x_bf, wk_bf, k_f, M, KV, Cc);
    gemm_bf16<<<dim3(KV / 64, M / 128), 256, 0, stream>>>(x_bf, wv_bf, v_f, M, KV, Cc);

    // 3) RMSNorm + RoPE -> bf16; V straight conversion
    rmsnorm_rope<<<M, 256, 0, stream>>>(q_f, cosb, sinb, qw, q_bf, Cc, 16);
    rmsnorm_rope<<<M, 256, 0, stream>>>(k_f, cosb, sinb, kw, k_bf, KV, 4);
    cvt(v_f, v_bf, M * KV);

    // 4) Flash attention (GQA 4:1)
    attn_wmma<<<dim3(Tc / 64, Bc * 16), 128, 0, stream>>>(q_bf, k_bf, v_bf, y_bf);

    // 5) Output projection, fp32 out
    gemm_bf16<<<dim3(Cc / 64, M / 128), 256, 0, stream>>>(y_bf, wo_bf, out, M, Cc, Cc);
}
